// PureSAGEConv_69209103008247
// MI455X (gfx1250) — matrix operand order repaired and checked by reading, not compiled, not run on hardware
//
#include <hip/hip_runtime.h>

#define N_NODES 50000
#define IN_CH   512
#define OUT_CH  512
#define N_EDGES 800000
#define KDIM    (2 * IN_CH)   // 1024

typedef __attribute__((ext_vector_type(16))) __bf16 v16bf;
typedef __attribute__((ext_vector_type(8)))  __bf16 v8bf;
typedef __attribute__((ext_vector_type(4)))  __bf16 v4bf;
typedef __attribute__((ext_vector_type(8)))  float  v8f;

__device__ __forceinline__ __bf16 f2bf(float f) { return (__bf16)f; }

// ---------------- zero scratch (sum + count) ----------------
__global__ void sage_zero_ws(float* __restrict__ ws, long n) {
    long i = (long)blockIdx.x * blockDim.x + threadIdx.x;
    long stride = (long)gridDim.x * blockDim.x;
    for (; i < n; i += stride) ws[i] = 0.0f;
}

// ---------------- gather x[src] and scatter-add into sum[dst] ----------------
// One 256-thread block per edge; each thread handles 2 consecutive channels.
// sum buffer (~102MB) largely resident in the 192MB L2 -> atomics resolve at L2.
__global__ __launch_bounds__(256) void sage_scatter(
    const float* __restrict__ x,
    const int*   __restrict__ src,
    const int*   __restrict__ dst,
    float* __restrict__ sum,
    float* __restrict__ cnt) {
    const int e = blockIdx.x;
    const int s = src[e];
    const int d = dst[e];
    if (threadIdx.x == 0) unsafeAtomicAdd(&cnt[d], 1.0f);
    const float* xs = x   + (size_t)s * IN_CH;
    float*       sd = sum + (size_t)d * IN_CH;
    const int c = threadIdx.x * 2;
    float2 v = *(const float2*)(xs + c);
    unsafeAtomicAdd(&sd[c + 0], v.x);
    unsafeAtomicAdd(&sd[c + 1], v.y);
}

// ---------------- pack h = concat(x, sum/max(cnt,1)) as bf16 ----------------
__global__ __launch_bounds__(256) void sage_pack_h(
    const float* __restrict__ x,
    const float* __restrict__ sum,
    const float* __restrict__ cnt,
    __bf16* __restrict__ hbf) {
    const long i4 = (long)blockIdx.x * blockDim.x + threadIdx.x;   // float4 index
    const long total4 = (long)N_NODES * IN_CH / 4;
    if (i4 >= total4) return;
    const long row = i4 / (IN_CH / 4);
    const int  c   = (int)(i4 % (IN_CH / 4)) * 4;
    float cn = cnt[row];
    cn = cn > 1.0f ? cn : 1.0f;
    const float inv = 1.0f / cn;
    const float4 xv = *(const float4*)(x   + row * IN_CH + c);
    const float4 sv = *(const float4*)(sum + row * IN_CH + c);
    v4bf xa, ma;
    xa[0] = f2bf(xv.x); xa[1] = f2bf(xv.y); xa[2] = f2bf(xv.z); xa[3] = f2bf(xv.w);
    ma[0] = f2bf(sv.x * inv); ma[1] = f2bf(sv.y * inv);
    ma[2] = f2bf(sv.z * inv); ma[3] = f2bf(sv.w * inv);
    __bf16* hr = hbf + row * KDIM;
    *(v4bf*)(hr + c)         = xa;
    *(v4bf*)(hr + IN_CH + c) = ma;
}

// ---------------- pack W as bf16 (2MB -> 1MB, L2 resident) ----------------
__global__ __launch_bounds__(256) void sage_pack_w(
    const float* __restrict__ W, __bf16* __restrict__ wbf) {
    const size_t i4 = (size_t)blockIdx.x * blockDim.x + threadIdx.x; // 131072 total
    const float4 wv = *(const float4*)(W + i4 * 4);
    v4bf p;
    p[0] = f2bf(wv.x); p[1] = f2bf(wv.y); p[2] = f2bf(wv.z); p[3] = f2bf(wv.w);
    *(v4bf*)(wbf + i4 * 4) = p;
}

// ---------------- out = h @ W.T + b via v_wmma_f32_16x16x32_bf16 ----------------
// One wave computes a 32x64 output block (2 M-tiles x 4 N-tiles, 8 accumulators).
// Per k-step: 2 A frags + 4 B frags feed 8 WMMAs. M tail: A reads clamp to the
// last row; stores use a UNIFORM scalar branch per 16-row block (N_NODES is a
// multiple of 16, so blocks never straddle the boundary) -> no per-lane EXEC
// juggling in the epilogue, and WMMAs always run with EXEC all-ones.
__global__ __launch_bounds__(128) void sage_gemm(
    const __bf16* __restrict__ hbf,
    const __bf16* __restrict__ wbf,
    const float*  __restrict__ bias,
    float* __restrict__ out) {
    const int lane = threadIdx.x & 31;
    const int wave = threadIdx.x >> 5;
    const int tile = blockIdx.x * 4 + wave;
    const int nt = tile & 7;            // 512 / 64 = 8 N-groups
    const int mt = tile >> 3;           // ceil(3125/2) = 1563 M-groups
    const int m0 = mt * 32;
    const int n0 = nt * 64;
    const int half = lane >> 4;         // K-group selector per ISA layout
    const int lr   = lane & 15;         // row (A) / col (B,D) within tile

    const __bf16* arow[2];
    #pragma unroll
    for (int i = 0; i < 2; ++i) {
        int m = m0 + i * 16 + lr;
        if (m > N_NODES - 1) m = N_NODES - 1;      // clamp tail reads
        arow[i] = hbf + (size_t)m * KDIM + half * 16;
    }
    const __bf16* brow[4];
    #pragma unroll
    for (int j = 0; j < 4; ++j)
        brow[j] = wbf + (size_t)(n0 + j * 16 + lr) * KDIM + half * 16;

    v8f acc[2][4] = {};
    for (int k0 = 0; k0 < KDIM; k0 += 32) {
        v16bf aF[2], bF[4];
        #pragma unroll
        for (int i = 0; i < 2; ++i) {
            // A: lane holds row, K = half*16 + {0..15} contiguous
            v8bf lo = *(const v8bf*)(arow[i] + k0);
            v8bf hi = *(const v8bf*)(arow[i] + k0 + 8);
            aF[i] = __builtin_shufflevector(lo, hi,
                0, 1, 2, 3, 4, 5, 6, 7, 8, 9, 10, 11, 12, 13, 14, 15);
        }
        #pragma unroll
        for (int j = 0; j < 4; ++j)
            // B: lane holds col, K = half*16 + {0..15} contiguous (32B)
            bF[j] = *(const v16bf*)(brow[j] + k0);

        #pragma unroll
        for (int i = 0; i < 2; ++i)
            #pragma unroll
            for (int j = 0; j < 4; ++j)
                acc[i][j] = __builtin_amdgcn_wmma_f32_16x16x32_bf16(
                    false, aF[i], false, bF[j],
                    (short)0, acc[i][j], false, false);
    }

    #pragma unroll
    for (int i = 0; i < 2; ++i) {
        // Uniform (scalar) validity test for this 16-row block: N_NODES % 16 == 0
        // means the block is entirely valid or entirely out of range.
        if (m0 + i * 16 < N_NODES) {
            const int mrow = m0 + i * 16 + half * 8;
            #pragma unroll
            for (int j = 0; j < 4; ++j) {
                const float bv = bias[n0 + j * 16 + lr];
                float* od = out + (size_t)mrow * OUT_CH + n0 + j * 16 + lr;
                #pragma unroll
                for (int q = 0; q < 8; ++q)
                    od[(size_t)q * OUT_CH] = acc[i][j][q] + bv;
            }
        }
    }
}

extern "C" void kernel_launch(void* const* d_in, const int* in_sizes, int n_in,
                              void* d_out, int out_size, void* d_ws, size_t ws_size,
                              hipStream_t stream) {
    const float* x    = (const float*)d_in[0];
    const int*   ei   = (const int*)  d_in[1];   // (2, E): row0 = src, row1 = dst
    const float* W    = (const float*)d_in[2];
    const float* bias = (const float*)d_in[3];
    float* out = (float*)d_out;

    // ws layout: hbf (N*KDIM bf16) | wbf (OUT*KDIM bf16) | sum (N*IN_CH f32) | cnt (N f32)
    __bf16* hbf = (__bf16*)d_ws;
    __bf16* wbf = hbf + (size_t)N_NODES * KDIM;
    float*  sum = (float*)(wbf + (size_t)OUT_CH * KDIM);
    float*  cnt = sum + (size_t)N_NODES * IN_CH;

    const long ztotal = (long)N_NODES * IN_CH + N_NODES;
    sage_zero_ws<<<4096, 256, 0, stream>>>(sum, ztotal);

    sage_pack_w<<<(OUT_CH * KDIM / 4) / 256, 256, 0, stream>>>(W, wbf);

    sage_scatter<<<N_EDGES, 256, 0, stream>>>(x, ei, ei + N_EDGES, sum, cnt);

    const int packh = (N_NODES * IN_CH / 4 + 255) / 256;   // 25000
    sage_pack_h<<<packh, 256, 0, stream>>>(x, sum, cnt, hbf);

    const int mgroups = (N_NODES / 16 + 1) / 2;            // 1563
    const int ngroups = OUT_CH / 64;                       // 8
    sage_gemm<<<(mgroups * ngroups) / 4, 128, 0, stream>>>(hbf, wbf, bias, out);
}
